// LSTM_encoder2_20547123544823
// MI455X (gfx1250) — matrix operand order repaired and checked by reading, not compiled, not run on hardware
//
#include <hip/hip_runtime.h>
#include <hip/hip_bf16.h>
#include <math.h>

#define VOCAB 20000
#define D     512
#define B     64
#define S     256
#define C     16
#define G4D   (4 * D)   // 2048
#define NBLK  32        // persistent recurrence blocks

// Optional gfx1250 async global->LDS copy (ASYNCcnt path). Guarded so the
// file still compiles if this toolchain doesn't declare the builtins.
#if defined(__has_builtin)
#if __has_builtin(__builtin_amdgcn_global_load_async_to_lds_b128) && \
    __has_builtin(__builtin_amdgcn_s_wait_asynccnt)
#define HAVE_ASYNC_LDS 1
#endif
#endif
#ifndef HAVE_ASYNC_LDS
#define HAVE_ASYNC_LDS 0
#endif

typedef __attribute__((ext_vector_type(16))) __bf16 bf16x16;
typedef __attribute__((ext_vector_type(8)))  __bf16 bf16x8;
typedef __attribute__((ext_vector_type(8)))  float  f32x8;
typedef int i32x4 __attribute__((vector_size(16)));

#if HAVE_ASYNC_LDS
typedef __attribute__((address_space(1))) i32x4* gas_i32x4p;   // non-const
typedef __attribute__((address_space(3))) i32x4* las_i32x4p;
#endif

// ---------------------------------------------------------------------------
// WMMA fragment loaders (v_wmma_f32_16x16x32_bf16 layouts, wave32)
// A tile 16x32 (MxK): lane L holds row (L&15); elems 0..7 = K[8*(L>>4) .. +7],
//                     elems 8..15 = same + 16.
// B tile 32x16 (KxN): lane L holds col (L&15); elems 0..7 = K[8*(L>>4) .. +7], 8..15 = same + 16.
// C/D tile:           elem r of lane L = (M = 8*(L>>4)+r, N = L&15).
// ---------------------------------------------------------------------------
__device__ __forceinline__ bf16x16 load_frag_a(const __bf16* p) {
    union { bf16x16 v; bf16x8 h[2]; } u;
    u.h[0] = *(const bf16x8*)(p);
    u.h[1] = *(const bf16x8*)(p + 16);
    return u.v;
}
__device__ __forceinline__ bf16x16 load_frag_b(const __bf16* p) {
    union { bf16x16 v; bf16x8 h[2]; } u;
    u.h[0] = *(const bf16x8*)(p);
    u.h[1] = *(const bf16x8*)(p + 16);
    return u.v;
}
__device__ __forceinline__ f32x8 wmma_bf16(bf16x16 a, bf16x16 b, f32x8 c) {
    return __builtin_amdgcn_wmma_f32_16x16x32_bf16(
        false, a, false, b, (short)0, c, false, false);
}
__device__ __forceinline__ float sigmoidf_(float x) {
    return 1.0f / (1.0f + expf(-x));
}

// ---------------------------------------------------------------------------
__global__ void k_f32_to_bf16(const float* __restrict__ src,
                              __bf16* __restrict__ dst, int n) {
    int i = blockIdx.x * blockDim.x + threadIdx.x;
    if (i < n) dst[i] = (__bf16)src[i];
}

// ---------------------------------------------------------------------------
// Embedding gather-sum -> x stored T-MAJOR: x[s*B + b, :] (bf16).
// ~512 MB of gathered reads -> ~22 us at 23.3 TB/s; memory-bound.
// ---------------------------------------------------------------------------
__global__ void k_embed(const int* __restrict__ seqs,
                        const float* __restrict__ table,
                        __bf16* __restrict__ xbf) {
    __shared__ int idx[C];
    int v = blockIdx.x;                 // b*S + s (seqs layout)
    if (threadIdx.x < C) idx[threadIdx.x] = seqs[v * C + threadIdx.x];
    __syncthreads();
    int b = v >> 8;                     // S = 256
    int s = v & 255;
    size_t xrow = (size_t)s * B + b;    // t-major
    int d0 = threadIdx.x;
    float s0 = 0.f, s1 = 0.f;
#pragma unroll
    for (int c = 0; c < C; ++c) {
        const float* row = table + (size_t)idx[c] * D;
        s0 += row[d0];
        s1 += row[d0 + 256];
    }
    xbf[xrow * D + d0]       = (__bf16)s0;
    xbf[xrow * D + d0 + 256] = (__bf16)s1;
}

// ---------------------------------------------------------------------------
// gx = X @ w_ih.T, 2x4 register-blocked (8 acc = 64 VGPRs + 6 frags = 48:
// fits the register budget -> no scratch spills, unlike 4x4).
// Output in C-fragment TILED layout: tile id = mt*128 + nt, 32 B/lane.
// ---------------------------------------------------------------------------
__global__ void __launch_bounds__(256)
k_gemm_gx(const __bf16* __restrict__ xbf,
          const __bf16* __restrict__ wih,
          float* __restrict__ gxt) {
    int grp  = blockIdx.x * (blockDim.x >> 5) + (threadIdx.x >> 5); // 0..16383
    int lane = threadIdx.x & 31;
    int ng = grp & 31;                   // n-group: 4 n-tiles
    int mg = grp >> 5;                   // m-group: 2 m-tiles (x is t-major)
    int mBase = mg * 32;
    int nBase = ng * 64;
    const int akoff = (lane >> 4) * 8;
    const int bkoff = (lane >> 4) * 8;

    const __bf16* ap[2];
    const __bf16* bp[4];
#pragma unroll
    for (int i = 0; i < 2; ++i)
        ap[i] = xbf + (size_t)(mBase + i * 16 + (lane & 15)) * D + akoff;
#pragma unroll
    for (int j = 0; j < 4; ++j)
        bp[j] = wih + (size_t)(nBase + j * 16 + (lane & 15)) * D + bkoff;

    f32x8 acc[2][4] = {};
#pragma unroll 1
    for (int kb = 0; kb < D / 32; ++kb) {
        bf16x16 af[2], bf[4];
#pragma unroll
        for (int i = 0; i < 2; ++i) af[i] = load_frag_a(ap[i] + kb * 32);
#pragma unroll
        for (int j = 0; j < 4; ++j) bf[j] = load_frag_b(bp[j] + kb * 32);
#pragma unroll
        for (int i = 0; i < 2; ++i)
#pragma unroll
            for (int j = 0; j < 4; ++j)
                acc[i][j] = wmma_bf16(af[i], bf[j], acc[i][j]);
    }
#pragma unroll
    for (int i = 0; i < 2; ++i)
#pragma unroll
        for (int j = 0; j < 4; ++j) {
            size_t tile = (size_t)(mg * 2 + i) * 128 + (ng * 4 + j);
            *(f32x8*)(gxt + (tile * 32 + lane) * 8) = acc[i][j];
        }
}

// ---------------------------------------------------------------------------
__global__ void k_init(__bf16* __restrict__ h0, __bf16* __restrict__ h1,
                       unsigned* __restrict__ bar) {
    int i = blockIdx.x * blockDim.x + threadIdx.x;   // B*D = 32768
    h0[i] = (__bf16)0.f;
    h1[i] = (__bf16)0.f;
    if (i < 2) bar[i] = 0u;
}

// ---------------------------------------------------------------------------
// Persistent LSTM recurrence: all 256 timesteps in ONE kernel.
// 32 blocks (= d-tiles) x 4 waves (= batch tiles). w_hh slice staged into LDS
// (async-to-LDS), cell state in VGPRs, gx consumed as WMMA C operand,
// h double-buffered with a device-wide barrier between steps.
// ---------------------------------------------------------------------------
__global__ void __launch_bounds__(128, 1)
k_lstm_recurrence(const __bf16* __restrict__ whh,
                  const float* __restrict__ gxt,
                  const int* __restrict__ lengths,
                  __bf16* __restrict__ hbuf0,
                  __bf16* __restrict__ hbuf1,
                  float* __restrict__ finalh,
                  unsigned* __restrict__ bar) {   // bar[0]=count, bar[1]=sense
    __shared__ __bf16 sw[4 * 16 * D];             // 64 KiB w_hh slice
    const int tid  = threadIdx.x;
    const int lane = tid & 31;
    const int mt   = tid >> 5;        // 0..3 batch tile
    const int dt   = blockIdx.x;      // 0..31 hidden-dim tile
    const int mBase = mt * 16;
    const int dBase = dt * 16;

    // stage w_hh rows {g*512 + dBase + c} x K=512 into LDS (64 rows x 1 KiB)
#if HAVE_ASYNC_LDS
    for (int i = tid; i < 64 * 64; i += 128) {    // 4096 x 16 B async copies
        int row = i >> 6;                         // g*16 + c
        int ch  = i & 63;
        int g = row >> 4, c = row & 15;
        __bf16* src = (__bf16*)(size_t)(whh + (size_t)(g * 512 + dBase + c) * D + ch * 8);
        __bf16* dst = sw + row * D + ch * 8;
        __builtin_amdgcn_global_load_async_to_lds_b128(
            (gas_i32x4p)(void*)src,
            (las_i32x4p)(void*)dst,
            0, 0);
    }
    __builtin_amdgcn_s_wait_asynccnt(0);
    __syncthreads();
#else
    for (int i = tid; i < 64 * 64; i += 128) {
        int row = i >> 6;                         // g*16 + c
        int ch  = i & 63;
        int g = row >> 4, c = row & 15;
        *(bf16x8*)(sw + row * D + ch * 8) =
            *(const bf16x8*)(whh + (size_t)(g * 512 + dBase + c) * D + ch * 8);
    }
    __syncthreads();
#endif

    const int m0 = mBase + 8 * (lane >> 4);
    const int d  = dBase + (lane & 15);
    int len[8];
#pragma unroll
    for (int r = 0; r < 8; ++r) len[r] = lengths[m0 + r];

    const int akoff   = (lane >> 4) * 8;
    const int aRowOff = (mBase + (lane & 15)) * D + akoff;
    const int bkoff   = (lane >> 4) * 8;
    const __bf16* sb0 = sw + (0 * 16 + (lane & 15)) * D + bkoff;
    const __bf16* sb1 = sw + (1 * 16 + (lane & 15)) * D + bkoff;
    const __bf16* sb2 = sw + (2 * 16 + (lane & 15)) * D + bkoff;
    const __bf16* sb3 = sw + (3 * 16 + (lane & 15)) * D + bkoff;

    f32x8 creg = {};                              // cell state in registers

    for (int t = 0; t < S; ++t) {
        const __bf16* hin  = (t & 1) ? hbuf1 : hbuf0;
        __bf16*       hout = (t & 1) ? hbuf0 : hbuf1;

        // C operands straight from tiled gx (32 B contiguous per lane)
        size_t tb = ((size_t)(t * 4 + mt) * 128 + dt) * 32 + lane;
        f32x8 acc0 = *(const f32x8*)(gxt + (tb                  ) * 8);
        f32x8 acc1 = *(const f32x8*)(gxt + (tb + (size_t)32 * 32) * 8);
        f32x8 acc2 = *(const f32x8*)(gxt + (tb + (size_t)64 * 32) * 8);
        f32x8 acc3 = *(const f32x8*)(gxt + (tb + (size_t)96 * 32) * 8);

        // prefetch next step's gx tiles (global_prefetch_b8)
        if (t + 1 < S) {
            size_t pn = ((size_t)((t + 1) * 4 + mt) * 128 + dt) * 32 + lane;
            __builtin_prefetch(gxt + (pn                  ) * 8, 0, 3);
            __builtin_prefetch(gxt + (pn + (size_t)32 * 32) * 8, 0, 3);
            __builtin_prefetch(gxt + (pn + (size_t)64 * 32) * 8, 0, 3);
            __builtin_prefetch(gxt + (pn + (size_t)96 * 32) * 8, 0, 3);
        }

        const __bf16* ap = hin + aRowOff;
#pragma unroll 4
        for (int kb = 0; kb < D / 32; ++kb) {
            bf16x16 a = load_frag_a(ap + kb * 32);
            acc0 = wmma_bf16(a, load_frag_b(sb0 + kb * 32), acc0);
            acc1 = wmma_bf16(a, load_frag_b(sb1 + kb * 32), acc1);
            acc2 = wmma_bf16(a, load_frag_b(sb2 + kb * 32), acc2);
            acc3 = wmma_bf16(a, load_frag_b(sb3 + kb * 32), acc3);
        }

        // gate math; lane exclusively owns (b = m0+r, d)
#pragma unroll
        for (int r = 0; r < 8; ++r) {
            float ig = sigmoidf_(acc0[r]);
            float fg = sigmoidf_(acc1[r]);
            float gg = tanhf(acc2[r]);
            float og = sigmoidf_(acc3[r]);
            float cn = fg * creg[r] + ig * gg;
            creg[r] = cn;
            float hn = og * tanhf(cn);
            hout[(size_t)(m0 + r) * D + d] = (__bf16)hn;
            if (len[r] - 1 == t)
                finalh[(m0 + r) * D + d] = hn;
        }

        // ---- device-wide barrier (sense reversal over NBLK blocks) ----
        __threadfence();
        __syncthreads();
        if (tid == 0) {
            unsigned my = __hip_atomic_load(&bar[1], __ATOMIC_RELAXED,
                                            __HIP_MEMORY_SCOPE_AGENT);
            if (__hip_atomic_fetch_add(&bar[0], 1u, __ATOMIC_ACQ_REL,
                                       __HIP_MEMORY_SCOPE_AGENT) == NBLK - 1u) {
                __hip_atomic_store(&bar[0], 0u, __ATOMIC_RELAXED,
                                   __HIP_MEMORY_SCOPE_AGENT);
                __hip_atomic_store(&bar[1], my + 1u, __ATOMIC_RELEASE,
                                   __HIP_MEMORY_SCOPE_AGENT);
            } else {
                while (__hip_atomic_load(&bar[1], __ATOMIC_ACQUIRE,
                                         __HIP_MEMORY_SCOPE_AGENT) == my)
                    __builtin_amdgcn_s_sleep(1);
            }
        }
        __syncthreads();
        __threadfence();
    }
}

// ---------------------------------------------------------------------------
__global__ void k_out(const float* __restrict__ finalh,
                      const float* __restrict__ wout,
                      const float* __restrict__ bout,
                      float* __restrict__ out) {
    int tid = threadIdx.x;            // 128 threads
    int b = tid >> 1, j = tid & 1;
    float s = bout[j];
    for (int d = 0; d < D; ++d)
        s += finalh[b * D + d] * wout[j * D + d];
    out[b * 2 + j] = s;
}

// ---------------------------------------------------------------------------
extern "C" void kernel_launch(void* const* d_in, const int* in_sizes, int n_in,
                              void* d_out, int out_size, void* d_ws, size_t ws_size,
                              hipStream_t stream) {
    const int*   seqs    = (const int*)d_in[0];
    const int*   lengths = (const int*)d_in[1];
    const float* table   = (const float*)d_in[2];
    const float* w_ih    = (const float*)d_in[3];
    const float* w_hh    = (const float*)d_in[4];
    const float* w_out   = (const float*)d_in[5];
    const float* b_out   = (const float*)d_in[6];
    float* out = (float*)d_out;

    char* ws = (char*)d_ws;
    size_t off = 0;
    __bf16*   xbf   = (__bf16*)(ws + off);   off += (size_t)B * S * D * 2;   // 16 MiB
    __bf16*   wihb  = (__bf16*)(ws + off);   off += (size_t)G4D * D * 2;     //  2 MiB
    __bf16*   whhb  = (__bf16*)(ws + off);   off += (size_t)G4D * D * 2;     //  2 MiB
    __bf16*   hbuf0 = (__bf16*)(ws + off);   off += (size_t)B * D * 2;       // 64 KiB
    __bf16*   hbuf1 = (__bf16*)(ws + off);   off += (size_t)B * D * 2;       // 64 KiB
    float*    fin   = (float*)(ws + off);    off += (size_t)B * D * 4;       //128 KiB
    unsigned* bar   = (unsigned*)(ws + off); off += 256;
    float*    gxt   = (float*)(ws + off);    off += (size_t)B * S * G4D * 4; //128 MiB

    int nw = G4D * D;
    k_f32_to_bf16<<<(nw + 255) / 256, 256, 0, stream>>>(w_ih, wihb, nw);
    k_f32_to_bf16<<<(nw + 255) / 256, 256, 0, stream>>>(w_hh, whhb, nw);

    k_embed<<<B * S, 256, 0, stream>>>(seqs, table, xbf);

    // 16384 waves (2x4-blocked), 8 waves/block -> 2048 blocks
    k_gemm_gx<<<2048, 256, 0, stream>>>(xbf, wihb, gxt);

    k_init<<<(B * D) / 256, 256, 0, stream>>>(hbuf0, hbuf1, bar);

    k_lstm_recurrence<<<NBLK, 128, 0, stream>>>(whhb, gxt, lengths,
                                                hbuf0, hbuf1, fin, bar);

    k_out<<<1, 128, 0, stream>>>(fin, w_out, b_out, out);
}
